// AttentionLayer_9955734192808
// MI455X (gfx1250) — compile-verified, matrix operand order checked by reading, not run on hardware
//
#include <hip/hip_runtime.h>

#define DECD 1024
#define ENC2D 2048
#define BSZ 64
#define LSZ 1024
#define MROWS (BSZ * LSZ)       // 65536 flat rows of enc_output

typedef __attribute__((ext_vector_type(16))) __bf16 v16bf;
typedef __attribute__((ext_vector_type(8)))  float  v8f;
typedef __attribute__((ext_vector_type(4)))  unsigned int u32x4;
typedef __attribute__((ext_vector_type(8)))  int i32x8;
typedef __attribute__((ext_vector_type(4)))  int i32x4;

union Frag { v16bf v; unsigned int u[8]; };

// split f32 -> bf16 hi (truncate) + bf16 lo (truncate of residual); hi+lo ~ x to 2^-16
__device__ __forceinline__ void split_pack(float x0, float x1,
                                           unsigned& hi, unsigned& lo) {
    unsigned h0 = __float_as_uint(x0) & 0xffff0000u;
    unsigned h1 = __float_as_uint(x1) & 0xffff0000u;
    hi = h1 | (h0 >> 16);
    float l0 = x0 - __uint_as_float(h0);
    float l1 = x1 - __uint_as_float(h1);
    lo = (__float_as_uint(l1) & 0xffff0000u) | (__float_as_uint(l0) >> 16);
}

// ---------------------------------------------------------------------------
// TDM: issue one Tensor-Data-Mover DMA of a 128x256 f32 tile (row stride 2048
// floats in memory) into LDS, inserting 4 pad dwords after every 256 dwords so
// the LDS image has the bank-conflict-avoiding 260-dword row stride.
// Tracked with TENSORcnt; one wave issues, workgroup syncs after the wait.
// ---------------------------------------------------------------------------
#define MT   128
#define KT   256
#define LDSS (KT + 4)            // dword stride; rows stay 16B aligned

__device__ __forceinline__ void tdm_issue(const float* gsrc, unsigned lds_off) {
    unsigned long long ga = (unsigned long long)(size_t)gsrc;
    u32x4 g0 = {
        1u,                                             // count=1, user mode
        lds_off,                                        // D#.lds_addr
        (unsigned)ga,                                   // global_addr[31:0]
        (unsigned)((ga >> 32) & 0x01ffffffull) | 0x80000000u  // addr[56:32], type=2
    };
    i32x8 g1 = {
        (int)((2u << 16) | (1u << 20) | (7u << 22) | (3u << 25)),
        //     data_size=4B  pad_en     every 256dw  pad 4 dwords
        (int)(2048u << 16),           // tensor_dim0 = 2048 (lo16)
        0,                            // tensor_dim0 hi16, tensor_dim1 lo16
        (int)(1u | (256u << 16)),     // tensor_dim1 = 65536 (hi16), tile_dim0 = 256
        128,                          // tile_dim1 = 128, tile_dim2 = 0
        2048,                         // tensor_dim0_stride lo32 (elements)
        0, 0
    };
    i32x4 gz4 = {0, 0, 0, 0};
#if defined(__clang_major__) && __clang_major__ >= 23
    i32x8 gz8 = {0, 0, 0, 0, 0, 0, 0, 0};
    __builtin_amdgcn_tensor_load_to_lds(g0, g1, gz4, gz4, gz8, 0);
#else
    __builtin_amdgcn_tensor_load_to_lds(g0, g1, gz4, gz4, 0);
#endif
}

// ---------------------------------------------------------------------------
// Kernel A: dbase[b,d] = b_score[d] + sum_k dec_hidden[b,k] * W_dec[k,d]
// ---------------------------------------------------------------------------
__global__ void decproj_kernel(const float* __restrict__ dec_hidden,
                               const float* __restrict__ Wsc,
                               const float* __restrict__ b_score,
                               float* __restrict__ dbase) {
    int b = blockIdx.x >> 2;
    int d = ((blockIdx.x & 3) << 8) + threadIdx.x;
    const float* drow = dec_hidden + b * DECD;
    const float* wcol = Wsc + d;                 // W_dec rows 0..1023, stride 1024
    float acc = b_score[d];
    for (int k = 0; k < DECD; k += 4) {
        acc += drow[k + 0] * wcol[(size_t)(k + 0) * DECD];
        acc += drow[k + 1] * wcol[(size_t)(k + 1) * DECD];
        acc += drow[k + 2] * wcol[(size_t)(k + 2) * DECD];
        acc += drow[k + 3] * wcol[(size_t)(k + 3) * DECD];
    }
    dbase[b * DECD + d] = acc;
}

// ---------------------------------------------------------------------------
// Kernel B: fused GEMM + tanh + v-dot -> scores[b,l]
//   X = enc_output as (65536, 2048); W_enc = W_score rows 1024..3071
//   scores[m] = sum_n v[n] * tanh(dbase[m&63, n] + (X @ W_enc)[m, n])
// 128 rows/workgroup, 8 waves; wave w owns N-block (h*32 + r*8 + w)*16.
// Split-bf16 (3x WMMA) ~ fp32 accuracy at bf16 WMMA throughput.
// X tiles double-buffered in LDS, streamed by the TDM and overlapped
// with WMMA compute via TENSORcnt software pipelining.
// ---------------------------------------------------------------------------
#define RND  4                   // N-blocks per wave per half
#define HALV 2
#define NKC  (ENC2D / KT)        // 8 K-chunks per half

__launch_bounds__(256, 2)
__global__ void score_kernel(const float* __restrict__ X,
                             const float* __restrict__ Wsc,
                             const float* __restrict__ dbase,
                             const float* __restrict__ vvec,
                             float* __restrict__ scores) {
    __shared__ float xs[2][MT * LDSS];
    __shared__ float scoreAcc[MT];

    const int tid   = threadIdx.x;
    const int wave  = tid >> 5;
    const int lane  = tid & 31;
    const int lhalf = lane >> 4;      // 0: lanes 0-15, 1: lanes 16-31
    const int l16   = lane & 15;
    const int m0    = blockIdx.x * MT;
    const float* Wenc = Wsc + (size_t)DECD * DECD;   // row 1024 onward
    const float* Xrow = X + (size_t)m0 * ENC2D;

    if (tid < MT) scoreAcc[tid] = 0.0f;

    // pipeline prologue: DMA first X chunk into buffer 0
    if (wave == 0) tdm_issue(Xrow, (unsigned)(size_t)&xs[0][0]);

    const v8f zacc = {0.f, 0.f, 0.f, 0.f, 0.f, 0.f, 0.f, 0.f};

    for (int h = 0; h < HALV; ++h) {
        v8f acc[8][RND];
        #pragma unroll
        for (int st = 0; st < 8; ++st)
            #pragma unroll
            for (int r = 0; r < RND; ++r)
                acc[st][r] = zacc;

        for (int kc = 0; kc < NKC; ++kc) {
            const int c = h * NKC + kc;           // global chunk index
            __syncthreads();                      // other buffer free for DMA
            if (wave == 0) {
                if (c + 1 < HALV * NKC) {
                    tdm_issue(Xrow + ((c + 1) % NKC) * KT,
                              (unsigned)(size_t)&xs[(c + 1) & 1][0]);
                    __builtin_amdgcn_s_wait_tensorcnt(1);  // chunk c landed
                } else {
                    __builtin_amdgcn_s_wait_tensorcnt(0);
                }
            }
            __syncthreads();                      // tile visible to all waves
            const float* xbuf = &xs[c & 1][0];

            for (int ks = 0; ks < KT / 32; ++ks) {
                const int kstep = kc * KT + ks * 32;
                // ---- B fragments: W_enc[kstep..+31, nblock*16..+15], split bf16
                Frag bhi[RND], blo[RND];
                #pragma unroll
                for (int r = 0; r < RND; ++r) {
                    const int nb = h * 32 + r * 8 + wave;
                    const int n  = nb * 16 + l16;
                    #pragma unroll
                    for (int j = 0; j < 8; ++j) {
                        const int p = ((j & 3) << 1) + ((j & 4) ? 16 : 0) + (lhalf << 3);
                        float w0 = Wenc[(size_t)(kstep + p)     * DECD + n];
                        float w1 = Wenc[(size_t)(kstep + p + 1) * DECD + n];
                        split_pack(w0, w1, bhi[r].u[j], blo[r].u[j]);
                    }
                }
                // ---- A fragments from LDS, reused across RND N-blocks
                #pragma unroll
                for (int st = 0; st < 8; ++st) {
                    Frag ahi, alo;
                    const int am = st * 16 + l16;
                    #pragma unroll
                    for (int j = 0; j < 8; ++j) {
                        const int p = ((j & 3) << 1) + ((j & 4) ? 16 : 0) + (lhalf << 3);
                        float2 x2 = *(const float2*)(xbuf + am * LDSS + ks * 32 + p);
                        split_pack(x2.x, x2.y, ahi.u[j], alo.u[j]);
                    }
                    #pragma unroll
                    for (int r = 0; r < RND; ++r) {
                        acc[st][r] = __builtin_amdgcn_wmma_f32_16x16x32_bf16(
                            false, ahi.v, false, bhi[r].v, (short)0, acc[st][r], false, false);
                        acc[st][r] = __builtin_amdgcn_wmma_f32_16x16x32_bf16(
                            false, alo.v, false, bhi[r].v, (short)0, acc[st][r], false, false);
                        acc[st][r] = __builtin_amdgcn_wmma_f32_16x16x32_bf16(
                            false, ahi.v, false, blo[r].v, (short)0, acc[st][r], false, false);
                    }
                }
            }
        }
        // ---- epilogue for this half: tanh, *v, reduce over N, accumulate rows
        #pragma unroll
        for (int st = 0; st < 8; ++st) {
            #pragma unroll
            for (int r = 0; r < RND; ++r) {
                const int nb = h * 32 + r * 8 + wave;
                const int n  = nb * 16 + l16;
                const float vn = vvec[n];
                #pragma unroll
                for (int vr = 0; vr < 8; ++vr) {
                    const int Mrow = st * 16 + vr + (lhalf << 3);
                    const int b    = Mrow & 63;          // m0 multiple of 128
                    float val = tanhf(acc[st][r][vr] + dbase[b * DECD + n]) * vn;
                    val += __shfl_xor(val, 1, 16);
                    val += __shfl_xor(val, 2, 16);
                    val += __shfl_xor(val, 4, 16);
                    val += __shfl_xor(val, 8, 16);
                    if (l16 == 0) atomicAdd(&scoreAcc[Mrow], val);
                }
            }
        }
    }
    __syncthreads();
    if (tid < MT) {
        int m = m0 + tid;                        // m = l*64 + b
        scores[(m & 63) * LSZ + (m >> 6)] = scoreAcc[tid];
    }
}

// ---------------------------------------------------------------------------
// Kernel C: softmax over L per batch row
// ---------------------------------------------------------------------------
__global__ void softmax_kernel(const float* __restrict__ scores,
                               float* __restrict__ alpha) {
    __shared__ float red[256];
    const int b = blockIdx.x, tid = threadIdx.x;
    const float* s = scores + b * LSZ;
    float mx = -1e30f;
    #pragma unroll
    for (int i = 0; i < 4; ++i) mx = fmaxf(mx, s[tid + i * 256]);
    red[tid] = mx; __syncthreads();
    for (int o = 128; o > 0; o >>= 1) {
        if (tid < o) red[tid] = fmaxf(red[tid], red[tid + o]);
        __syncthreads();
    }
    mx = red[0]; __syncthreads();
    float e[4], sum = 0.f;
    #pragma unroll
    for (int i = 0; i < 4; ++i) { e[i] = __expf(s[tid + i * 256] - mx); sum += e[i]; }
    red[tid] = sum; __syncthreads();
    for (int o = 128; o > 0; o >>= 1) {
        if (tid < o) red[tid] += red[tid + o];
        __syncthreads();
    }
    float inv = 1.0f / red[0];
    #pragma unroll
    for (int i = 0; i < 4; ++i) alpha[b * LSZ + tid + i * 256] = e[i] * inv;
}

// ---------------------------------------------------------------------------
// Kernel D: attention[b,e] = sum_l alpha[b,l] * enc_output[l,b,e]
// grid split over (b, e-chunk, l-chunk); f32 global atomics into zeroed out.
// Prefetch next strided enc row (global_prefetch_b8) to hide the 512KB stride.
// ---------------------------------------------------------------------------
__global__ void attn_kernel(const float* __restrict__ enc,
                            const float* __restrict__ alpha,
                            float* __restrict__ out) {
    const int b  = blockIdx.x & 63;
    const int ec = (blockIdx.x >> 6) & 7;
    const int lc = blockIdx.x >> 9;
    const int e  = (ec << 8) + threadIdx.x;
    const int l0 = lc << 7;
    float accv = 0.f;
    for (int l = l0; l < l0 + 128; ++l) {
        const int lp = (l + 4) & (LSZ - 1);
        __builtin_prefetch(enc + ((size_t)(lp << 6) + b) * ENC2D + e, 0, 1);
        accv += alpha[b * LSZ + l] * enc[((size_t)(l << 6) + b) * ENC2D + e];
    }
    atomicAdd(out + b * ENC2D + e, accv);
}

// ---------------------------------------------------------------------------
extern "C" void kernel_launch(void* const* d_in, const int* in_sizes, int n_in,
                              void* d_out, int out_size, void* d_ws, size_t ws_size,
                              hipStream_t stream) {
    const float* dec_hidden = (const float*)d_in[0];
    const float* enc_output = (const float*)d_in[1];
    const float* W_score    = (const float*)d_in[2];
    const float* b_score    = (const float*)d_in[3];
    const float* v          = (const float*)d_in[4];
    float* out = (float*)d_out;

    float* ws     = (float*)d_ws;
    float* dbase  = ws;             // 64*1024
    float* scores = ws + 65536;     // 64*1024
    float* alpha  = ws + 131072;    // 64*1024

    hipMemsetAsync(d_out, 0, sizeof(float) * (size_t)out_size, stream);
    decproj_kernel<<<256, 256, 0, stream>>>(dec_hidden, W_score, b_score, dbase);
    score_kernel<<<MROWS / MT, 256, 0, stream>>>(enc_output, W_score, dbase, v, scores);
    softmax_kernel<<<64, 256, 0, stream>>>(scores, alpha);
    attn_kernel<<<4096, 256, 0, stream>>>(enc_output, alpha, out);
}
